// PocketContextSurrogateModel_23802708755003
// MI455X (gfx1250) — matrix operand order repaired and checked by reading, not compiled, not run on hardware
//
#include <hip/hip_runtime.h>
#include <stdint.h>

#define HD __device__ __forceinline__

typedef __attribute__((ext_vector_type(16))) __bf16 v16bf;
typedef __attribute__((ext_vector_type(8)))  float  v8f;

// ---- model constants ----
#define B_     16
#define NL_    32
#define NP_    192
#define N_     224            // nodes per graph
#define H_     128
#define L_     4
#define NRBF_  16
#define TDIM_  64
#define NNODE  (B_*N_)        // 3584
#define EPG    12736          // edges per graph
#define E_TOT  (B_*EPG)       // 203776 (divisible by 32)
#define MSGIN  401
#define KPAD1  416            // 13 * 32
#define MT     32             // edge rows per block (two 16-row WMMA tiles)
#define STEP_  (4.0f/(NRBF_-1))
#define GAMMA_ (1.0f/(STEP_*STEP_))

HD unsigned short f2bf(float f) {
  union { float f; uint32_t u; } v; v.f = f;
  uint32_t u = v.u;
  u += 0x7FFFu + ((u >> 16) & 1u);      // round-to-nearest-even
  return (unsigned short)(u >> 16);
}
HD float silu_f(float x) { return x / (1.0f + __expf(-x)); }

// ---- analytic edge topology (matches _build_edges) ----
HD void edge_decode(int e, int* src, int* dst, int* et) {
  int g  = e / EPG;
  int le = e - g * EPG;
  int s, d, t;
  if (le < 2*NL_) {
    t = 0;
    if (le < NL_) { s = le;            d = (le + 1) & (NL_-1); }
    else          { int k = le - NL_;  s = (k + 1) & (NL_-1); d = k; }
  } else if (le < 2*NL_ + 2*NP_) {
    t = 1;
    int k = le - 2*NL_;
    if (k < NP_) { s = NL_ + k;                  d = NL_ + ((k + 1) % NP_); }
    else         { k -= NP_; s = NL_ + ((k + 1) % NP_); d = NL_ + k; }
  } else if (le < 2*NL_ + 2*NP_ + NL_*NP_) {
    t = 2;
    int k = le - (2*NL_ + 2*NP_);
    s = NL_ + (k >> 5); d = k & (NL_-1);         // pr -> lr
  } else {
    t = 3;
    int k = le - (2*NL_ + 2*NP_ + NL_*NP_);
    s = k & (NL_-1); d = NL_ + (k >> 5);         // lr -> pr
  }
  int base = g * N_;
  *src = base + s; *dst = base + d; *et = t;
}

// ---- block LayerNorm over H_=128 (one value per thread) ----
HD float block_ln(float x, float g, float bb, float* red, int c) {
  red[c] = x; __syncthreads();
  for (int s = 64; s > 0; s >>= 1) { if (c < s) red[c] += red[c + s]; __syncthreads(); }
  float m = red[0] * (1.0f / H_); __syncthreads();
  float d = x - m;
  red[c] = d * d; __syncthreads();
  for (int s = 64; s > 0; s >>= 1) { if (c < s) red[c] += red[c + s]; __syncthreads(); }
  float v = red[0] * (1.0f / H_); __syncthreads();
  return d * rsqrtf(v + 1e-5f) * g + bb;
}

// ================= weight prep: f32 row-major -> bf16 col-major (K padded) ==========
__global__ void wconv_kernel(const float* __restrict__ msg_w1, const float* __restrict__ msg_w2,
                             unsigned short* __restrict__ W1, unsigned short* __restrict__ W2) {
  int idx = blockIdx.x * blockDim.x + threadIdx.x;
  const int n1 = L_ * H_ * KPAD1;
  const int n2 = L_ * H_ * H_;
  if (idx < n1) {
    int l = idx / (H_ * KPAD1);
    int r = idx - l * (H_ * KPAD1);
    int n = r / KPAD1;
    int k = r - n * KPAD1;
    float v = (k < MSGIN) ? msg_w1[((size_t)l * MSGIN + k) * H_ + n] : 0.0f;
    W1[idx] = f2bf(v);
  } else if (idx < n1 + n2) {
    int j = idx - n1;
    int l = j / (H_ * H_);
    int r = j - l * (H_ * H_);
    int n = r / H_;
    int k = r - n * H_;
    W2[j] = f2bf(msg_w2[((size_t)l * H_ + k) * H_ + n]);
  }
}

// ================= time embedding MLP: te[B][H] =====================================
__global__ __launch_bounds__(128) void te_kernel(const float* __restrict__ t,
    const float* __restrict__ w1, const float* __restrict__ b1,
    const float* __restrict__ w2, const float* __restrict__ b2, float* __restrict__ te) {
  __shared__ float emb[TDIM_];
  __shared__ float z1[H_];
  int b = blockIdx.x, c = threadIdx.x;
  if (c < TDIM_ / 2) {
    float f = __expf(-logf(10000.0f) / (float)(TDIM_/2 - 1) * (float)c);
    float a = t[b] * f;
    emb[c] = __sinf(a); emb[c + TDIM_/2] = __cosf(a);
  }
  __syncthreads();
  float z = b1[c];
  for (int k = 0; k < TDIM_; ++k) z += emb[k] * w1[k * H_ + c];
  z1[c] = silu_f(z);
  __syncthreads();
  float o = b2[c];
  for (int k = 0; k < H_; ++k) o += z1[k] * w2[k * H_ + c];
  te[b * H_ + c] = o;
}

// ================= input features -> h0 (+coords_flat) ==============================
__global__ __launch_bounds__(128) void init_kernel(
    const float* __restrict__ lig, const float* __restrict__ poc,
    const float* __restrict__ w_in, const float* __restrict__ b_in,
    const float* __restrict__ node_emb, const float* __restrict__ lng,
    const float* __restrict__ lnb, const float* __restrict__ te,
    float* __restrict__ h, float* __restrict__ coords) {
  __shared__ float red[H_];
  int node = blockIdx.x, c = threadIdx.x;
  int g = node / N_, i = node - g * N_;
  float cx, cy, pos, nn; int type;
  if (i < NL_) { cx = lig[((size_t)g*NL_ + i)*2]; cy = lig[((size_t)g*NL_ + i)*2 + 1];
                 type = 1; pos = (float)i; nn = (float)NL_; }
  else { int j = i - NL_;
         cx = poc[((size_t)g*NP_ + j)*2]; cy = poc[((size_t)g*NP_ + j)*2 + 1];
         type = 0; pos = (float)j; nn = (float)NP_; }
  float ang = 6.2831853071795864769f * pos / nn;
  if (c == 0) { coords[node*2] = cx; coords[node*2 + 1] = cy; }
  float x = cx * w_in[0*H_ + c] + cy * w_in[1*H_ + c]
          + __sinf(ang) * w_in[2*H_ + c] + __cosf(ang) * w_in[3*H_ + c]
          + b_in[c] + node_emb[type*H_ + c] + te[g*H_ + c];
  h[(size_t)node*H_ + c] = block_ln(x, lng[c], lnb[c], red, c);
}

// ================= zero scratch =====================================================
__global__ void zero_kernel(float* __restrict__ p, int n) {
  int i = blockIdx.x * blockDim.x + threadIdx.x;
  if (i < n) p[i] = 0.0f;
}

// ================= edge message MLP (WMMA bf16) + scatter-add =======================
// 1 block = 32-edge tile (two 16-row WMMA M-tiles sharing B-fragments).
// 8 waves, wave w owns output columns [16w, 16w+16).
__global__ __launch_bounds__(256) void msg_kernel(
    const float* __restrict__ h, const float* __restrict__ coords,
    const float* __restrict__ edge_emb,
    const unsigned short* __restrict__ W1, const float* __restrict__ b1,
    const unsigned short* __restrict__ W2, const float* __restrict__ b2,
    float* __restrict__ agg, int layer) {
  __shared__ unsigned short tileA[MT * KPAD1];   // msg_in tile, bf16
  __shared__ unsigned short tileM[MT * H_];      // hidden tile, bf16
  __shared__ int   sSrc[MT], sDst[MT], sEt[MT];
  __shared__ float sDist[MT];

  const int tid  = threadIdx.x;
  const int lane = tid & 31;
  const int wave = tid >> 5;
  const int e0   = blockIdx.x * MT;

  if (tid < MT) {
    int s, d, t; edge_decode(e0 + tid, &s, &d, &t);
    sSrc[tid] = s; sDst[tid] = d; sEt[tid] = t;
    float dx = coords[2*s]     - coords[2*d];
    float dy = coords[2*s + 1] - coords[2*d + 1];
    sDist[tid] = sqrtf(dx*dx + dy*dy);
  }
  __syncthreads();

  // Stage msg_in = [h[src] | h[dst] | edge_emb[etype] | radial | dist | pad] as bf16
  for (int idx = tid; idx < MT * KPAD1; idx += 256) {
    int r = idx / KPAD1;
    int c = idx - r * KPAD1;
    float v;
    if      (c < H_)          v = h[(size_t)sSrc[r]*H_ + c];
    else if (c < 2*H_)        v = h[(size_t)sDst[r]*H_ + (c - H_)];
    else if (c < 3*H_)        v = edge_emb[((size_t)layer*4 + sEt[r])*H_ + (c - 2*H_)];
    else if (c < 3*H_+NRBF_)  { float dd = sDist[r] - (float)(c - 3*H_) * STEP_;
                                v = __expf(-GAMMA_ * dd * dd); }
    else if (c == 3*H_+NRBF_) v = sDist[r];
    else                      v = 0.0f;
    tileA[idx] = f2bf(v);
  }
  __syncthreads();

  const int mrow  = lane & 15;
  const int rbase = (lane >> 4) << 3;   // accumulator row base: 0 or 8
  const int kA    = (lane >> 4) << 3;   // A-frag K base within step: 0 or 8
  const int kB    = (lane >> 4) << 3;   // B-frag K base within step: 0 or 8
  const int nCol  = (wave << 4) + (lane & 15);

  union Frag { uint4 u[2]; v16bf v; };

  // ---- GEMM1: [32 x 416] @ [416 x 128], B-frag shared by both M-tiles ----
  v8f acc0 = {0.f,0.f,0.f,0.f,0.f,0.f,0.f,0.f};
  v8f acc1 = {0.f,0.f,0.f,0.f,0.f,0.f,0.f,0.f};
  {
    const unsigned short* a0 = &tileA[mrow * KPAD1 + kA];
    const unsigned short* a1 = &tileA[(mrow + 16) * KPAD1 + kA];
    const unsigned short* bp = W1 + ((size_t)layer * H_ + nCol) * KPAD1 + kB;
    Frag fb, fa0, fa1, fbN, fa0N, fa1N;
    fb.u[0]  = *(const uint4*)(bp);       fb.u[1]  = *(const uint4*)(bp + 16);
    fa0.u[0] = *(const uint4*)(a0);       fa0.u[1] = *(const uint4*)(a0 + 16);
    fa1.u[0] = *(const uint4*)(a1);       fa1.u[1] = *(const uint4*)(a1 + 16);
    #pragma unroll
    for (int ks = 0; ks < KPAD1 / 32; ++ks) {
      if (ks + 1 < KPAD1 / 32) {          // prefetch next K-step before WMMAs
        int o = (ks + 1) * 32;
        fbN.u[0]  = *(const uint4*)(bp + o);      fbN.u[1]  = *(const uint4*)(bp + o + 16);
        fa0N.u[0] = *(const uint4*)(a0 + o);      fa0N.u[1] = *(const uint4*)(a0 + o + 16);
        fa1N.u[0] = *(const uint4*)(a1 + o);      fa1N.u[1] = *(const uint4*)(a1 + o + 16);
      }
      acc0 = __builtin_amdgcn_wmma_f32_16x16x32_bf16(false, fa0.v, false, fb.v,
                                                     (short)0, acc0, false, false);
      acc1 = __builtin_amdgcn_wmma_f32_16x16x32_bf16(false, fa1.v, false, fb.v,
                                                     (short)0, acc1, false, false);
      if (ks + 1 < KPAD1 / 32) { fb = fbN; fa0 = fa0N; fa1 = fa1N; }
    }
  }
  float bias1 = b1[layer * H_ + nCol];
  #pragma unroll
  for (int i = 0; i < 8; ++i) {
    tileM[(rbase + i) * H_ + nCol]      = f2bf(silu_f(acc0[i] + bias1));
    tileM[(16 + rbase + i) * H_ + nCol] = f2bf(silu_f(acc1[i] + bias1));
  }
  __syncthreads();

  // ---- GEMM2: [32 x 128] @ [128 x 128] ----
  v8f c0 = {0.f,0.f,0.f,0.f,0.f,0.f,0.f,0.f};
  v8f c1 = {0.f,0.f,0.f,0.f,0.f,0.f,0.f,0.f};
  {
    const unsigned short* a0 = &tileM[mrow * H_ + kA];
    const unsigned short* a1 = &tileM[(mrow + 16) * H_ + kA];
    const unsigned short* bp = W2 + ((size_t)layer * H_ + nCol) * H_ + kB;
    Frag fb, fa0, fa1, fbN, fa0N, fa1N;
    fb.u[0]  = *(const uint4*)(bp);       fb.u[1]  = *(const uint4*)(bp + 16);
    fa0.u[0] = *(const uint4*)(a0);       fa0.u[1] = *(const uint4*)(a0 + 16);
    fa1.u[0] = *(const uint4*)(a1);       fa1.u[1] = *(const uint4*)(a1 + 16);
    #pragma unroll
    for (int ks = 0; ks < H_ / 32; ++ks) {
      if (ks + 1 < H_ / 32) {
        int o = (ks + 1) * 32;
        fbN.u[0]  = *(const uint4*)(bp + o);      fbN.u[1]  = *(const uint4*)(bp + o + 16);
        fa0N.u[0] = *(const uint4*)(a0 + o);      fa0N.u[1] = *(const uint4*)(a0 + o + 16);
        fa1N.u[0] = *(const uint4*)(a1 + o);      fa1N.u[1] = *(const uint4*)(a1 + o + 16);
      }
      c0 = __builtin_amdgcn_wmma_f32_16x16x32_bf16(false, fa0.v, false, fb.v,
                                                   (short)0, c0, false, false);
      c1 = __builtin_amdgcn_wmma_f32_16x16x32_bf16(false, fa1.v, false, fb.v,
                                                   (short)0, c1, false, false);
      if (ks + 1 < H_ / 32) { fb = fbN; fa0 = fa0N; fa1 = fa1N; }
    }
  }

  // ---- epilogue: bias + SiLU, merge equal-dst runs, atomic scatter-add ----
  float bias2 = b2[layer * H_ + nCol];
  float vals[8];
  #pragma unroll
  for (int half = 0; half < 2; ++half) {
    const v8f& cc = half ? c1 : c0;
    int rb = half * 16 + rbase;
    #pragma unroll
    for (int i = 0; i < 8; ++i) vals[i] = silu_f(cc[i] + bias2);
    int i = 0;
    while (i < 8) {                       // rows with identical dst -> one atomic
      int d = sDst[rb + i];
      float s = vals[i];
      int j = i + 1;
      while (j < 8 && sDst[rb + j] == d) { s += vals[j]; ++j; }
      __hip_atomic_fetch_add(&agg[(size_t)d * H_ + nCol], s,
                             __ATOMIC_RELAXED, __HIP_MEMORY_SCOPE_AGENT);
      i = j;
    }
  }
}

// ================= node update MLP + residual LN ====================================
__global__ __launch_bounds__(128) void update_kernel(
    float* __restrict__ h, const float* __restrict__ agg,
    const float* __restrict__ uw1, const float* __restrict__ ub1,
    const float* __restrict__ uw2, const float* __restrict__ ub2,
    const float* __restrict__ lng, const float* __restrict__ lnb, int layer) {
  __shared__ float cat[2 * H_];
  __shared__ float u1s[H_];
  __shared__ float red[H_];
  int node = blockIdx.x, c = threadIdx.x;
  int gi = node % N_;
  float inv_deg = (gi < NL_) ? (1.0f / (2.0f + NP_)) : (1.0f / (2.0f + NL_));
  float hv = h[(size_t)node * H_ + c];
  cat[c]      = hv;
  cat[H_ + c] = agg[(size_t)node * H_ + c] * inv_deg;
  __syncthreads();
  float z = ub1[layer * H_ + c];
  for (int k = 0; k < 2 * H_; ++k) z += cat[k] * uw1[((size_t)layer * 2*H_ + k) * H_ + c];
  u1s[c] = silu_f(z);
  __syncthreads();
  float u = ub2[layer * H_ + c];
  for (int k = 0; k < H_; ++k) u += u1s[k] * uw2[((size_t)layer * H_ + k) * H_ + c];
  float x = hv + u;
  h[(size_t)node * H_ + c] = block_ln(x, lng[layer*H_ + c], lnb[layer*H_ + c], red, c);
}

// ================= ligand pooling + head MLP ========================================
__global__ __launch_bounds__(128) void head_kernel(
    const float* __restrict__ h, const float* __restrict__ w1, const float* __restrict__ b1,
    const float* __restrict__ w2, const float* __restrict__ b2, float* __restrict__ out) {
  __shared__ float pool[H_];
  __shared__ float red[H_];
  int b = blockIdx.x, c = threadIdx.x;
  float s = 0.0f;
  for (int i = 0; i < NL_; ++i) s += h[((size_t)b * N_ + i) * H_ + c];
  pool[c] = s * (1.0f / NL_);
  __syncthreads();
  float z = b1[c];
  for (int k = 0; k < H_; ++k) z += pool[k] * w1[k * H_ + c];
  red[c] = silu_f(z) * w2[c];
  __syncthreads();
  for (int s2 = 64; s2 > 0; s2 >>= 1) { if (c < s2) red[c] += red[c + s2]; __syncthreads(); }
  if (c == 0) out[b] = red[0] + b2[0];
}

// ====================================================================================
extern "C" void kernel_launch(void* const* d_in, const int* in_sizes, int n_in,
                              void* d_out, int out_size, void* d_ws, size_t ws_size,
                              hipStream_t stream) {
  const float* lig      = (const float*)d_in[0];
  const float* poc      = (const float*)d_in[1];
  const float* t        = (const float*)d_in[2];
  const float* w_in     = (const float*)d_in[3];
  const float* b_in     = (const float*)d_in[4];
  const float* node_emb = (const float*)d_in[5];
  const float* ln_in_g  = (const float*)d_in[6];
  const float* ln_in_b  = (const float*)d_in[7];
  const float* time_w1  = (const float*)d_in[8];
  const float* time_b1  = (const float*)d_in[9];
  const float* time_w2  = (const float*)d_in[10];
  const float* time_b2  = (const float*)d_in[11];
  const float* edge_emb = (const float*)d_in[12];
  const float* msg_w1   = (const float*)d_in[13];
  const float* msg_b1   = (const float*)d_in[14];
  const float* msg_w2   = (const float*)d_in[15];
  const float* msg_b2   = (const float*)d_in[16];
  const float* upd_w1   = (const float*)d_in[17];
  const float* upd_b1   = (const float*)d_in[18];
  const float* upd_w2   = (const float*)d_in[19];
  const float* upd_b2   = (const float*)d_in[20];
  const float* ln_g     = (const float*)d_in[21];
  const float* ln_b     = (const float*)d_in[22];
  const float* head_w1  = (const float*)d_in[23];
  const float* head_b1  = (const float*)d_in[24];
  const float* head_w2  = (const float*)d_in[25];
  const float* head_b2  = (const float*)d_in[26];
  float* out = (float*)d_out;

  char* ws = (char*)d_ws;
  size_t off = 0;
  auto wsalloc = [&](size_t bytes) -> void* {
    void* p = ws + off;
    off = (off + bytes + 255) & ~(size_t)255;
    return p;
  };
  float* h      = (float*)wsalloc((size_t)NNODE * H_ * 4);
  float* agg    = (float*)wsalloc((size_t)NNODE * H_ * 4);
  float* te     = (float*)wsalloc((size_t)B_ * H_ * 4);
  float* coords = (float*)wsalloc((size_t)NNODE * 2 * 4);
  unsigned short* W1 = (unsigned short*)wsalloc((size_t)L_ * H_ * KPAD1 * 2);
  unsigned short* W2 = (unsigned short*)wsalloc((size_t)L_ * H_ * H_ * 2);

  int nconv = L_ * H_ * KPAD1 + L_ * H_ * H_;
  wconv_kernel<<<(nconv + 255) / 256, 256, 0, stream>>>(msg_w1, msg_w2, W1, W2);
  te_kernel<<<B_, H_, 0, stream>>>(t, time_w1, time_b1, time_w2, time_b2, te);
  init_kernel<<<NNODE, H_, 0, stream>>>(lig, poc, w_in, b_in, node_emb,
                                        ln_in_g, ln_in_b, te, h, coords);
  for (int l = 0; l < L_; ++l) {
    zero_kernel<<<(NNODE * H_ + 255) / 256, 256, 0, stream>>>(agg, NNODE * H_);
    msg_kernel<<<E_TOT / MT, 256, 0, stream>>>(h, coords, edge_emb,
                                               W1, msg_b1, W2, msg_b2, agg, l);
    update_kernel<<<NNODE, H_, 0, stream>>>(h, agg, upd_w1, upd_b1,
                                            upd_w2, upd_b2, ln_g, ln_b, l);
  }
  head_kernel<<<B_, H_, 0, stream>>>(h, head_w1, head_b1, head_w2, head_b2, out);
}
